// DilatedGCN_38671885533577
// MI455X (gfx1250) — matrix operand order repaired and checked by reading, not compiled, not run on hardware
//
#include <hip/hip_runtime.h>

typedef __attribute__((ext_vector_type(16))) _Float16 v16h;
typedef __attribute__((ext_vector_type(8)))  _Float16 v8h;
typedef __attribute__((ext_vector_type(8)))  float    v8f;

#define B_   4
#define N_   2000
#define T_   12
#define CD   64            // C == D == 64
#define K_   3
#define DEG_ 8
#define E_   (N_*DEG_)     // 16000
#define M_   (B_*N_*T_)    // 96000 rows, (b,n,t)-major everywhere

// ---------------------------------------------------------------------------
// Prep:
//   Wt1[n][k] = f16( W_mlp[k][n] )                       (B-fragment layout)
//   Wc        = W_mlp @ W_attn[0:64]  (composed linear for the attention
//               logits; h_dst half + b_attn are per-segment constants that
//               cancel in the softmax), Wt2[n][k] = f16( Wc[k][n] )
//   bc        = b_mlp @ W_attn[0:64]
// ---------------------------------------------------------------------------
__global__ __launch_bounds__(256)
void prep_weights_kernel(const float* __restrict__ Wm, const float* __restrict__ Wa,
                         const float* __restrict__ bm,
                         _Float16* __restrict__ Wt1, _Float16* __restrict__ Wt2,
                         float* __restrict__ bc) {
    const int idx = blockIdx.x * blockDim.x + threadIdx.x;
    if (idx < CD * CD) {
        const int n = idx >> 6, k = idx & 63;
        Wt1[idx] = (_Float16)Wm[k * CD + n];
        float acc = 0.f;
#pragma unroll 8
        for (int d = 0; d < CD; ++d) acc += Wm[k * CD + d] * Wa[d * CD + n];
        Wt2[idx] = (_Float16)acc;
        if (idx < CD) {
            float bacc = 0.f;
#pragma unroll 8
            for (int d = 0; d < CD; ++d) bacc += bm[d] * Wa[d * CD + idx];
            bc[idx] = bacc;
        }
    }
}

// ---------------------------------------------------------------------------
// Fused dual GEMM: one wave owns a 16-row tile of X, loads the f16 A-fragments
// once, and produces BOTH outputs (feats = X@Wm+bm, S1 = X@Wc+bc) with
// 2 matrices x 4 N-tiles x 2 K-steps = 16 v_wmma_f32_16x16x32_f16 per wave.
// Row order is identity (b,n,t)-major -> zero-arithmetic epilogue.
// ---------------------------------------------------------------------------
__global__ __launch_bounds__(256)
void gemm64x2_wmma_kernel(const float* __restrict__ X,
                          const _Float16* __restrict__ Wt1, const float* __restrict__ b1,
                          const _Float16* __restrict__ Wt2, const float* __restrict__ b2,
                          float* __restrict__ Y1, float* __restrict__ Y2)
{
    const int lane = threadIdx.x & 31;
    const int wave = threadIdx.x >> 5;
    const int row0 = (blockIdx.x * 8 + wave) * 16;
    if (row0 >= M_) return;                // wave-uniform -> EXEC all-1s for WMMA
    const int l16 = lane & 15;
    const int hi  = lane >> 4;

    // A fragments (ISA 16-bit A 16x32 layout), f32 -> f16 convert in regs.
    const float* xr = X + (size_t)(row0 + l16) * CD;
    v16h a[2];
#pragma unroll
    for (int s = 0; s < 2; ++s) {
        const int k0 = s * 32 + hi * 8;    // halves 0..7  -> K = k0..k0+7
        const int k1 = k0 + 16;            // halves 8..15 -> K = k1..k1+7
#pragma unroll
        for (int i = 0; i < 8; ++i) {
            a[s][i]     = (_Float16)xr[k0 + i];
            a[s][i + 8] = (_Float16)xr[k1 + i];
        }
    }

    const _Float16* Wts[2] = {Wt1, Wt2};
    const float*    bs[2]  = {b1, b2};
    float*          Ys[2]  = {Y1, Y2};
#pragma unroll
    for (int m = 0; m < 2; ++m) {
        const _Float16* Wt   = Wts[m];
        const float*    bias = bs[m];
        float*          Y    = Ys[m];
#pragma unroll
        for (int nt = 0; nt < 4; ++nt) {
            const int nb = nt * 16;
            v8f acc = {0.f, 0.f, 0.f, 0.f, 0.f, 0.f, 0.f, 0.f};
#pragma unroll
            for (int s = 0; s < 2; ++s) {
                // B fragment: lane = column nb+l16, K = s*32 + hi*8 + 0..7, then +16
                const _Float16* wp = Wt + (size_t)(nb + l16) * CD + s * 32 + hi * 8;
                v8h blo = *(const v8h*)wp;
                v8h bhi = *(const v8h*)(wp + 16);
                v16h bfrag;
#pragma unroll
                for (int i = 0; i < 8; ++i) { bfrag[i] = blo[i]; bfrag[i + 8] = bhi[i]; }
                acc = __builtin_amdgcn_wmma_f32_16x16x32_f16(
                          false, a[s], false, bfrag, (short)0, acc, false, false);
            }
            const float bv = bias[nb + l16];
            // D layout: VGPR v -> row (v + hi*8), col (nb + l16); identity rows
#pragma unroll
            for (int v = 0; v < 8; ++v)
                Y[(size_t)(row0 + v + hi * 8) * CD + nb + l16] = acc[v] + bv;
        }
    }
}

// ---------------------------------------------------------------------------
// Edge stage: one wave per (b,n,t) token, 2 channels per lane. Softmax
// segments are the contiguous 8-edge runs 8n..8n+7 (dst = repeat(arange(N),8)).
// Gathers land in L2 (feats+S1 = 49MB < 192MB). Per-channel softmax over the
// 8 in-edges' S1 logits, weighted feats sum, x dilation weight, leaky + res.
// ---------------------------------------------------------------------------
__global__ __launch_bounds__(256)
void edge_attn_kernel(const float* __restrict__ feats, const float* __restrict__ S1,
                      const int* __restrict__ edges, const float* __restrict__ kw,
                      const float* __restrict__ Xin, float* __restrict__ Yout)
{
    const int lane = threadIdx.x & 31;
    const int tok  = (int)((blockIdx.x * blockDim.x + threadIdx.x) >> 5);
    if (tok >= M_) return;
    const int rem = tok % (N_ * T_);
    const int n   = rem / T_;
    const int c   = lane * 2;
    const long long base = (long long)tok * CD + c;     // row of (b,n,t)
    const float w0 = kw[0], w1 = kw[1], w2 = kw[2];
    const float wk[K_] = {w0, w1, w2};

    float ax = 0.f, ay = 0.f;
#pragma unroll
    for (int k = 0; k < K_; ++k) {
        const int* sp = edges + (size_t)k * 2 * E_ + n * DEG_;  // edges[k][0][8n..]
        const int4 i0 = *(const int4*)sp;                        // 16B-aligned
        const int4 i1 = *(const int4*)(sp + 4);
        const int idxs[DEG_] = {i0.x, i0.y, i0.z, i0.w, i1.x, i1.y, i1.z, i1.w};

        float sx[DEG_], sy[DEG_], fx[DEG_], fy[DEG_];
#pragma unroll
        for (int j = 0; j < DEG_; ++j) {
            // row(b, src, t) = row(b,n,t) + (src-n)*T*CD
            const long long off = base + (long long)(idxs[j] - n) * (T_ * CD);
            const float2 sv = *(const float2*)(S1 + off);
            const float2 fv = *(const float2*)(feats + off);
            sx[j] = sv.x; sy[j] = sv.y; fx[j] = fv.x; fy[j] = fv.y;
        }
        float mx = sx[0], my = sy[0];
#pragma unroll
        for (int j = 1; j < DEG_; ++j) { mx = fmaxf(mx, sx[j]); my = fmaxf(my, sy[j]); }
        float dx = 0.f, dy = 0.f, nx = 0.f, ny = 0.f;
#pragma unroll
        for (int j = 0; j < DEG_; ++j) {
            const float ex = __expf(sx[j] - mx);
            const float ey = __expf(sy[j] - my);
            dx += ex; dy += ey;
            nx += ex * fx[j]; ny += ey * fy[j];
        }
        ax += wk[k] * nx / dx;
        ay += wk[k] * ny / dy;
    }
    ax = (ax > 0.f) ? ax : 0.01f * ax;     // leaky_relu(0.01)
    ay = (ay > 0.f) ? ay : 0.01f * ay;
    const float2 rv = *(const float2*)(Xin + base);
    float2 ov; ov.x = ax + rv.x; ov.y = ay + rv.y;
    *(float2*)(Yout + base) = ov;
}

// ---------------------------------------------------------------------------
extern "C" void kernel_launch(void* const* d_in, const int* in_sizes, int n_in,
                              void* d_out, int out_size, void* d_ws, size_t ws_size,
                              hipStream_t stream) {
    (void)in_sizes; (void)n_in; (void)out_size; (void)ws_size;
    const float* Xin   = (const float*)d_in[0];   // [B,N,T,C] f32
    const float* Wmlp  = (const float*)d_in[1];   // [64,64]
    const float* bmlp  = (const float*)d_in[2];   // [64]
    const float* Wattn = (const float*)d_in[3];   // [128,64] (rows 0..63 used)
    // d_in[4] = b_attn: per-segment additive constant -> cancels in softmax
    const float* kw    = (const float*)d_in[5];   // [K,1]
    const int*   edges = (const int*)d_in[6];     // [K,2,E] int32
    float* out = (float*)d_out;

    // Workspace: feats (24.6MB) | S1 (24.6MB) | Wt1 (8KB) | Wt2 (8KB) | bc (256B)
    float*    feats = (float*)d_ws;
    float*    S1    = feats + (size_t)M_ * CD;
    _Float16* Wt1   = (_Float16*)(S1 + (size_t)M_ * CD);
    _Float16* Wt2   = Wt1 + CD * CD;
    float*    bc    = (float*)(Wt2 + CD * CD);

    hipLaunchKernelGGL(prep_weights_kernel, dim3(16), dim3(256), 0, stream,
                       Wmlp, Wattn, bmlp, Wt1, Wt2, bc);

    // feats = X@W_mlp + b_mlp  AND  S1 = X@(W_mlp@W1) + b_mlp@W1, one pass
    const int blocks_gemm = (M_ / 16) / 8;   // 750
    hipLaunchKernelGGL(gemm64x2_wmma_kernel, dim3(blocks_gemm), dim3(256), 0, stream,
                       Xin, Wt1, bmlp, Wt2, bc, feats, S1);

    const int blocks_edge = M_ / 8;          // 12000, 8 wave-tokens per block
    hipLaunchKernelGGL(edge_attn_kernel, dim3(blocks_edge), dim3(256), 0, stream,
                       feats, S1, edges, kw, Xin, out);
}
